// enhance_mask_79817672229083
// MI455X (gfx1250) — matrix operand order repaired and checked by reading, not compiled, not run on hardware
//
#include <hip/hip_runtime.h>
#include <hip/hip_bf16.h>

// ---------------------------------------------------------------------------
// Problem constants (from reference setup_inputs)
// ---------------------------------------------------------------------------
#define BB   4
#define CC   16    // Cx == C1
#define CCA  8
#define NN   1024
#define DD   64    // D == L
#define LL   64
#define TOPK 5

#define BETA        0.02f
#define INV_SQRT10  0.3162277660168379f

typedef __attribute__((ext_vector_type(16))) __bf16 v16bf;
typedef __attribute__((ext_vector_type(8)))  float  v8f;

union BF16Frag {
    v16bf v;
    unsigned int u[8];
};

// ---------------------------------------------------------------------------
// WMMA fragment loaders (bf16, 16x16x32, wave32) — contiguous k order chosen here (A, B match)
// A 16x32: lanes 0-15 hold M=lane, K = kb+0..15 contiguous
//          lanes 16-31 hold M=lane-16, K = kb+16..31 contiguous
// B 32x16: lanes 0-15 hold N=lane,  K = kb+0..15 contiguous
//          lanes 16-31 hold N=lane-16, K = kb+16..31 contiguous
// For a Gram matmul (B = X^T) both loads are contiguous 16B/32B row chunks.
// ---------------------------------------------------------------------------
static __device__ inline v16bf load_afrag(const __bf16* rowbase, int lane, int kb) {
    const int half = lane >> 4, lr = lane & 15;
    const __bf16* p = rowbase + lr * DD;
    const unsigned int* p0 = (const unsigned int*)(p + kb + half * 16);
    const unsigned int* p1 = (const unsigned int*)(p + kb + 8 + half * 16);
    BF16Frag f;
#pragma unroll
    for (int i = 0; i < 4; ++i) { f.u[i] = p0[i]; f.u[4 + i] = p1[i]; }
    return f.v;
}

static __device__ inline v16bf load_bfrag(const __bf16* base, int m0, int lane, int kb) {
    const int half = lane >> 4, lr = lane & 15;
    const unsigned int* p = (const unsigned int*)(base + (size_t)(m0 + lr) * DD + kb + half * 16);
    BF16Frag f;
#pragma unroll
    for (int i = 0; i < 8; ++i) f.u[i] = p[i];
    return f.v;
}

static __device__ inline v8f wmma_bf16(v16bf a, v16bf b, v8f c) {
    return __builtin_amdgcn_wmma_f32_16x16x32_bf16(false, a, false, b, (short)0, c, false, false);
}

// ---------------------------------------------------------------------------
// Kernel 0: f32 -> bf16 conversion for x and mask (one pass, 4M elems each)
// ---------------------------------------------------------------------------
__global__ void k_convert(const float* __restrict__ x, const float* __restrict__ mask,
                          __bf16* __restrict__ xbf, __bf16* __restrict__ mbf, int n) {
    int i = blockIdx.x * blockDim.x + threadIdx.x;
    if (i < n) {
        xbf[i] = (__bf16)x[i];
        mbf[i] = (__bf16)mask[i];
    }
}

// ---------------------------------------------------------------------------
// Kernel 1: S[b,n,m] = (1/16) * sum_c softmax_m( x_bc[n,:] . x_bc[m,:] )
// One block = one (b, 16-row tile). 8 waves x 8 column tiles = 1024 cols.
// 64KB LDS row buffer; softmax via shfl reductions over 16-lane row groups.
// ---------------------------------------------------------------------------
__global__ __launch_bounds__(256) void k_gram_softmax(const __bf16* __restrict__ xbf,
                                                      float* __restrict__ Ssum) {
    __shared__ float Srows[16 * NN];   // 64 KB

    const int blk  = blockIdx.x;            // 0..255
    const int b    = blk >> 6;              // /64
    const int n0   = (blk & 63) << 4;       // *16
    const int tid  = threadIdx.x;
    const int lane = tid & 31;
    const int wave = tid >> 5;
    const int half = lane >> 4, lr = lane & 15;
    const int r    = tid >> 4;              // owned row (0..15)
    const int sub  = tid & 15;              // 16 threads per row

    float acc[64];
#pragma unroll
    for (int j = 0; j < 64; ++j) acc[j] = 0.0f;

    for (int c = 0; c < CC; ++c) {
        const __bf16* xc = xbf + ((size_t)(b * CC + c)) * NN * DD;
        const v16bf a0 = load_afrag(xc + (size_t)n0 * DD, lane, 0);
        const v16bf a1 = load_afrag(xc + (size_t)n0 * DD, lane, 32);

#pragma unroll
        for (int i = 0; i < 8; ++i) {
            const int mt = wave * 8 + i;           // column tile index (0..63)
            const v16bf b0 = load_bfrag(xc, mt * 16, lane, 0);
            const v16bf b1 = load_bfrag(xc, mt * 16, lane, 32);
            v8f ct = {};
            ct = wmma_bf16(a0, b0, ct);
            ct = wmma_bf16(a1, b1, ct);
#pragma unroll
            for (int j = 0; j < 8; ++j)
                Srows[(j + half * 8) * NN + mt * 16 + lr] = ct[j];
        }
        __syncthreads();

        // row softmax: each thread owns cols sub + 16*j of row r
        float vals[64];
        float mx = -INFINITY;
#pragma unroll
        for (int j = 0; j < 64; ++j) {
            vals[j] = Srows[r * NN + sub + 16 * j];
            mx = fmaxf(mx, vals[j]);
        }
#pragma unroll
        for (int s = 1; s < 16; s <<= 1) mx = fmaxf(mx, __shfl_xor(mx, s, 32));
        float sum = 0.0f;
#pragma unroll
        for (int j = 0; j < 64; ++j) { vals[j] = __expf(vals[j] - mx); sum += vals[j]; }
#pragma unroll
        for (int s = 1; s < 16; s <<= 1) sum += __shfl_xor(sum, s, 32);
        const float inv = 0.0625f / sum;   // softmax normalize * (1/16 channel mean)
#pragma unroll
        for (int j = 0; j < 64; ++j) acc[j] += vals[j] * inv;
        __syncthreads();
    }

    float* out = Ssum + ((size_t)b * NN + n0 + r) * NN;
#pragma unroll
    for (int j = 0; j < 64; ++j) out[sub + 16 * j] = acc[j];
}

// ---------------------------------------------------------------------------
// Kernel 2: per channel c:
//   mm = mask_bc . mask_bc^T (WMMA)            -> LDS
//   A_out = mean_ca(A) + BETA*sigmoid(S*mm/sqrt(10))  -> global (write-once)
//   top-5 per row (5 shfl-argmax rounds), gather mask rows, max -> mask_updated
// ---------------------------------------------------------------------------
__global__ __launch_bounds__(256) void k_mask_out(const __bf16* __restrict__ mbf,
                                                  const float* __restrict__ mask_f32,
                                                  const float* __restrict__ Afull,
                                                  const float* __restrict__ Ssum,
                                                  float* __restrict__ Aout,
                                                  float* __restrict__ MaskUpd) {
    __shared__ float Srows[16 * NN];   // 64 KB

    const int blk  = blockIdx.x;
    const int b    = blk >> 6;
    const int n0   = (blk & 63) << 4;
    const int tid  = threadIdx.x;
    const int lane = tid & 31;
    const int wave = tid >> 5;
    const int half = lane >> 4, lr = lane & 15;
    const int r    = tid >> 4;
    const int sub  = tid & 15;

    // S (pre-scaled by 1/sqrt(10)) and channel-mean of A, ownership layout
    float sreg[64], amean[64];
    {
        const float* sp = Ssum + ((size_t)b * NN + n0 + r) * NN;
#pragma unroll
        for (int j = 0; j < 64; ++j) sreg[j] = sp[sub + 16 * j] * INV_SQRT10;
#pragma unroll
        for (int j = 0; j < 64; ++j) amean[j] = 0.0f;
        for (int ca = 0; ca < CCA; ++ca) {
            const float* ap = Afull + (((size_t)(b * CCA + ca)) * NN + n0 + r) * NN;
#pragma unroll
            for (int j = 0; j < 64; ++j) amean[j] += ap[sub + 16 * j];
        }
#pragma unroll
        for (int j = 0; j < 64; ++j) amean[j] *= 0.125f;
    }

    for (int c = 0; c < CC; ++c) {
        const __bf16* mc = mbf + ((size_t)(b * CC + c)) * NN * LL;
        const v16bf a0 = load_afrag(mc + (size_t)n0 * LL, lane, 0);
        const v16bf a1 = load_afrag(mc + (size_t)n0 * LL, lane, 32);

#pragma unroll
        for (int i = 0; i < 8; ++i) {
            const int mt = wave * 8 + i;
            const v16bf b0 = load_bfrag(mc, mt * 16, lane, 0);
            const v16bf b1 = load_bfrag(mc, mt * 16, lane, 32);
            v8f ct = {};
            ct = wmma_bf16(a0, b0, ct);
            ct = wmma_bf16(a1, b1, ct);
#pragma unroll
            for (int j = 0; j < 8; ++j)
                Srows[(j + half * 8) * NN + mt * 16 + lr] = ct[j];
        }
        __syncthreads();

        // fuse: A_out = Amean + BETA*sigmoid(S*mm/sqrt(10)); overwrite LDS in place
        float* aop = Aout + (((size_t)(b * CC + c)) * NN + n0 + r) * NN;
#pragma unroll
        for (int j = 0; j < 64; ++j) {
            const int col = sub + 16 * j;
            const float mm = Srows[r * NN + col];
            const float w  = 1.0f / (1.0f + __expf(-sreg[j] * mm));
            const float a  = amean[j] + BETA * w;
            Srows[r * NN + col] = a;   // each thread re-reads only its own cells
            aop[col] = a;              // A_out streamed out exactly once
        }

        // top-5 per row: 5 argmax rounds, shfl-reduced across the 16 row lanes
        int picked[TOPK];
#pragma unroll
        for (int t5 = 0; t5 < TOPK; ++t5) {
            float bestv = -INFINITY;
            int   besti = 0x7fffffff;
            for (int j = 0; j < 64; ++j) {
                const int col = sub + 16 * j;
                bool skip = false;
                for (int q = 0; q < t5; ++q) skip |= (picked[q] == col);
                const float v = Srows[r * NN + col];
                if (!skip && (v > bestv || (v == bestv && col < besti))) {
                    bestv = v; besti = col;
                }
            }
#pragma unroll
            for (int s = 1; s < 16; s <<= 1) {
                const float ov = __shfl_xor(bestv, s, 32);
                const int   oi = __shfl_xor(besti, s, 32);
                if (ov > bestv || (ov == bestv && oi < besti)) { bestv = ov; besti = oi; }
            }
            picked[t5] = besti;
        }

        // gather top-5 mask rows (f32 originals), max-reduce; 4 L-elems per thread
        {
            const float* mrow = mask_f32 + ((size_t)(b * CC + c)) * NN * LL;
            float o[4];
#pragma unroll
            for (int q = 0; q < TOPK; ++q) {
                const float* g = mrow + (size_t)picked[q] * LL + sub * 4;
                if (q == 0) {
#pragma unroll
                    for (int i = 0; i < 4; ++i) o[i] = g[i];
                } else {
#pragma unroll
                    for (int i = 0; i < 4; ++i) o[i] = fmaxf(o[i], g[i]);
                }
            }
            float* mo = MaskUpd + (((size_t)(b * CC + c)) * NN + n0 + r) * LL + sub * 4;
#pragma unroll
            for (int i = 0; i < 4; ++i) mo[i] = o[i];
        }
        __syncthreads();
    }
}

// ---------------------------------------------------------------------------
// Launch
// ---------------------------------------------------------------------------
extern "C" void kernel_launch(void* const* d_in, const int* in_sizes, int n_in,
                              void* d_out, int out_size, void* d_ws, size_t ws_size,
                              hipStream_t stream) {
    const float* x    = (const float*)d_in[0];   // (4,16,1024,64)
    const float* A    = (const float*)d_in[1];   // (4,8,1024,1024)
    const float* mask = (const float*)d_in[2];   // (4,16,1024,64)
    // d_in[3] = k (ignored; reference hard-codes top-5)

    float* Aout    = (float*)d_out;                              // (4,16,1024,1024)
    float* MaskUpd = (float*)d_out + (size_t)BB * CC * NN * NN;  // (4,16,1024,64)

    // workspace: xbf (8MB) | mbf (8MB) | Ssum (16MB)  => 32 MB
    __bf16* xbf = (__bf16*)d_ws;
    __bf16* mbf = xbf + (size_t)BB * CC * NN * DD;
    float*  Ssum = (float*)(mbf + (size_t)BB * CC * NN * LL);

    const int nconv = BB * CC * NN * DD;   // 4,194,304
    k_convert<<<(nconv + 255) / 256, 256, 0, stream>>>(x, mask, xbf, mbf, nconv);

    const int nblocks = BB * (NN / 16);    // 256
    k_gram_softmax<<<nblocks, 256, 0, stream>>>(xbf, Ssum);
    k_mask_out<<<nblocks, 256, 0, stream>>>(mbf, mask, A, Ssum, Aout, MaskUpd);
}
